// SlotAttention_18184891531791
// MI455X (gfx1250) — matrix operand order repaired and checked by reading, not compiled, not run on hardware
//
#include <hip/hip_runtime.h>
#include <hip/hip_bf16.h>
#include <math.h>

// ---------------------------------------------------------------------------
// Types / helpers
// ---------------------------------------------------------------------------
typedef __attribute__((ext_vector_type(16))) __bf16 v16bf;
typedef __attribute__((ext_vector_type(8)))  float  v8f;
#if __has_builtin(__builtin_amdgcn_cvt_pk_bf16_f32)
typedef __attribute__((ext_vector_type(2))) __bf16 v2bf;
#endif

__device__ inline v8f vzero() {
    v8f z;
#pragma unroll
    for (int i = 0; i < 8; ++i) z[i] = 0.0f;
    return z;
}

// Fast packed f32x2 -> bf16x2 in one VALU op.
__device__ inline unsigned packbf2(float lo, float hi) {
#if __has_builtin(__builtin_amdgcn_cvt_pk_bf16_f32)
    union { v2bf v; unsigned u; } r;
    r.v = __builtin_amdgcn_cvt_pk_bf16_f32(lo, hi);
    return r.u;
#else
    // byte-select the two high halves: out = {hi[3],hi[2],lo[3],lo[2]} (truncate)
    return __builtin_amdgcn_perm(__float_as_uint(hi), __float_as_uint(lo), 0x07060302u);
#endif
}
__device__ inline unsigned short f2bf(float f) {
    return (unsigned short)(packbf2(f, 0.0f) & 0xFFFFu);
}
__device__ inline float bf2f(unsigned short h) {
    return __uint_as_float(((unsigned)h) << 16);
}

__device__ inline v8f wmma_bf16(v16bf a, v16bf b, v8f c) {
    return __builtin_amdgcn_wmma_f32_16x16x32_bf16(
        /*neg_a=*/false, a, /*neg_b=*/false, b,
        /*c_mod=*/(short)0, c, /*reuse_a=*/false, /*reuse_b=*/false);
}

// A fragment (16x32, bf16) from row-major bf16 source. ISA 7.12.2:
//   lane: m = lane&15, half = lane>>4 ; VGPR vg holds K = 16*(vg>>2)+8*half+2*(vg&3)+{0,1}
__device__ inline v16bf frag_a_bf16(const unsigned short* p, long row0, int ld, int k0) {
    const int lane = threadIdx.x & 31, half = lane >> 4, m = lane & 15;
    const unsigned short* rp = p + (row0 + m) * (long)ld + k0 + 8 * half;
    union { v16bf v; unsigned u[8]; } r;
#pragma unroll
    for (int vg = 0; vg < 8; ++vg) {
        int k = 16 * (vg >> 2) + 2 * (vg & 3);
        r.u[vg] = *(const unsigned*)(rp + k);
    }
    return r.v;
}

// Same A layout but from an f32 row-major source, converting (and scaling) to bf16.
__device__ inline v16bf frag_a_f32(const float* p, long row0, int ld, int k0, float mul) {
    const int lane = threadIdx.x & 31, half = lane >> 4, m = lane & 15;
    const float* rp = p + (row0 + m) * (long)ld + k0 + 8 * half;
    union { v16bf v; unsigned u[8]; } r;
#pragma unroll
    for (int vg = 0; vg < 8; ++vg) {
        int k = 16 * (vg >> 2) + 2 * (vg & 3);
        float2 f = *(const float2*)(rp + k);
        r.u[vg] = packbf2(f.x * mul, f.y * mul);
    }
    return r.v;
}

// B fragment (32x16, bf16) from a *transposed-stored* matrix T where T[col][k]
// is row-major with leading dim ldt. Dense analogue of ISA sparse-B layout:
//   lane: n = lane&15 (column), half = lane>>4 ; VGPR vg holds K = 16*(vg>>2)+8*half+2*(vg&3)+{0,1}
__device__ inline v16bf frag_b(const unsigned short* T, long col0, int ldt, int k0) {
    const int lane = threadIdx.x & 31, half = lane >> 4, n = lane & 15;
    const unsigned short* rp = T + (col0 + n) * (long)ldt + k0 + 8 * half;
    union { v16bf v; unsigned u[8]; } r;
#pragma unroll
    for (int vg = 0; vg < 8; ++vg) r.u[vg] = *(const unsigned*)(rp + 16 * (vg >> 2) + 2 * (vg & 3));
    return r.v;
}

// Prefetch `bytes` of a buffer across the whole block (lowers to global_prefetch_b8).
__device__ inline void block_prefetch(const void* p, int bytes) {
    const char* c = (const char*)p;
    for (int off = threadIdx.x * 128; off < bytes; off += 256 * 128)
        __builtin_prefetch(c + off, 0, 3);
}

// ---------------------------------------------------------------------------
// Kernel 0: weight conversion f32 -> bf16 (optionally transposed)
// src [R][C] f32 ; transpose ? dst[C][R] : dst[R][C]
// ---------------------------------------------------------------------------
__global__ void k_wconv(const float* __restrict__ src, unsigned short* __restrict__ dst,
                        int R, int C, int transpose) {
    long total = (long)R * C;
    for (long i = (long)blockIdx.x * blockDim.x + threadIdx.x; i < total;
         i += (long)gridDim.x * blockDim.x) {
        int r = (int)(i / C), c = (int)(i % C);
        unsigned short v = f2bf(src[i]);
        if (transpose) dst[(long)c * R + r] = v; else dst[i] = v;
    }
}

// ---------------------------------------------------------------------------
// Kernel: slots = mu + sigma * noise   (slots lives in d_out, f32 [32][16][256])
// ---------------------------------------------------------------------------
__global__ void k_slots_init(const float* __restrict__ mu, const float* __restrict__ sig,
                             const float* __restrict__ noise, float* __restrict__ slots) {
    int i = blockIdx.x * 256 + threadIdx.x;
    if (i < 32 * 16 * 256) { int j = i & 4095; slots[i] = mu[j] + sig[j] * noise[i]; }
}

// ---------------------------------------------------------------------------
// Kernel 1: x = LN(inputs@W_in + b_in); k = x@Wk+bk (row-major bf16);
//           v = x@Wv+bv stored transposed [b][d][n] bf16.
// Grid: 1024 blocks of 256 threads; each block = 64 rows of (B*N).
// ---------------------------------------------------------------------------
__global__ __launch_bounds__(256) void k_xkv(
    const float* __restrict__ inputs, const float* __restrict__ b_in,
    const float* __restrict__ ln_g, const float* __restrict__ ln_b,
    const unsigned short* __restrict__ Wint,  // [256][512] bf16 (W_in^T)
    const unsigned short* __restrict__ Wkt,   // [256][256] bf16 (Wk^T)
    const unsigned short* __restrict__ Wvt,   // [256][256] bf16 (Wv^T)
    const float* __restrict__ bk, const float* __restrict__ bv,
    unsigned short* __restrict__ kbf,         // [B*N][256] bf16
    unsigned short* __restrict__ vtbf)        // [B][256][2048] bf16
{
    __shared__ unsigned short xbf[64 * 256];          // 32 KB
    __shared__ float stat[4][16][2][2];               // [rowtile][row][colhalf][sum,sumsq]
    const int tid = threadIdx.x, lane = tid & 31, w = tid >> 5;
    const int rt = w >> 1, ch = w & 1;
    const int half = lane >> 4, nIdx = lane & 15;
    const long rowBase = (long)blockIdx.x * 64 + rt * 16;

    // warm L2 for the stage-2 weights while stage 1 runs
    block_prefetch(Wkt, 256 * 256 * 2);
    block_prefetch(Wvt, 256 * 256 * 2);

    // ---- stage 1: t = inputs @ W_in ----
    v8f acc[8];
#pragma unroll
    for (int t = 0; t < 8; ++t) acc[t] = vzero();
    for (int kb = 0; kb < 512; kb += 32) {
        v16bf a = frag_a_f32(inputs, rowBase, 512, kb, 1.0f);
#pragma unroll
        for (int t = 0; t < 8; ++t)
            acc[t] = wmma_bf16(a, frag_b(Wint, ch * 128 + t * 16, 512, kb), acc[t]);
    }
    // bias + per-row stats (rows live in one 16-lane half; reduce with shfl_xor)
    float ps[8], pq[8];
#pragma unroll
    for (int i = 0; i < 8; ++i) { ps[i] = 0.f; pq[i] = 0.f; }
#pragma unroll
    for (int t = 0; t < 8; ++t) {
        int c = ch * 128 + t * 16 + nIdx;
        float bb = b_in[c];
#pragma unroll
        for (int i = 0; i < 8; ++i) {
            acc[t][i] += bb;
            ps[i] += acc[t][i];
            pq[i] += acc[t][i] * acc[t][i];
        }
    }
#pragma unroll
    for (int i = 0; i < 8; ++i)
        for (int off = 1; off < 16; off <<= 1) {
            ps[i] += __shfl_xor(ps[i], off);
            pq[i] += __shfl_xor(pq[i], off);
        }
    if (nIdx == 0) {
#pragma unroll
        for (int i = 0; i < 8; ++i) {
            stat[rt][8 * half + i][ch][0] = ps[i];
            stat[rt][8 * half + i][ch][1] = pq[i];
        }
    }
    __syncthreads();
    float mu_[8], rs_[8];
#pragma unroll
    for (int i = 0; i < 8; ++i) {
        int m = 8 * half + i;
        float s0 = stat[rt][m][0][0] + stat[rt][m][1][0];
        float q0 = stat[rt][m][0][1] + stat[rt][m][1][1];
        float mu = s0 * (1.0f / 256.0f);
        float var = q0 * (1.0f / 256.0f) - mu * mu;
        mu_[i] = mu; rs_[i] = rsqrtf(var + 1e-5f);
    }
#pragma unroll
    for (int t = 0; t < 8; ++t) {
        int c = ch * 128 + t * 16 + nIdx;
        float g = ln_g[c], bt = ln_b[c];
#pragma unroll
        for (int i = 0; i < 8; ++i) {
            int m = 8 * half + i;
            xbf[(rt * 16 + m) * 256 + c] = f2bf((acc[t][i] - mu_[i]) * rs_[i] * g + bt);
        }
    }
    __syncthreads();

    // ---- stage 2: k = x@Wk, v = x@Wv ----
    v8f ak[8], av[8];
#pragma unroll
    for (int t = 0; t < 8; ++t) { ak[t] = vzero(); av[t] = vzero(); }
    for (int kb = 0; kb < 256; kb += 32) {
        v16bf a = frag_a_bf16(xbf, rt * 16, 256, kb);
#pragma unroll
        for (int t = 0; t < 8; ++t) {
            int c0 = ch * 128 + t * 16;
            ak[t] = wmma_bf16(a, frag_b(Wkt, c0, 256, kb), ak[t]);
            av[t] = wmma_bf16(a, frag_b(Wvt, c0, 256, kb), av[t]);
        }
    }
    const int bIdx  = (int)(((long)blockIdx.x * 64) / 2048);
    const int nLoc0 = (int)(((long)blockIdx.x * 64) % 2048) + rt * 16 + 8 * half;
#pragma unroll
    for (int t = 0; t < 8; ++t) {
        int c = ch * 128 + t * 16 + nIdx;
        float bkc = bk[c], bvc = bv[c];
#pragma unroll
        for (int i = 0; i < 8; ++i)
            kbf[(rowBase + 8 * half + i) * 256 + c] = f2bf(ak[t][i] + bkc);
        uint4 u;
        u.x = packbf2(av[t][0] + bvc, av[t][1] + bvc);
        u.y = packbf2(av[t][2] + bvc, av[t][3] + bvc);
        u.z = packbf2(av[t][4] + bvc, av[t][5] + bvc);
        u.w = packbf2(av[t][6] + bvc, av[t][7] + bvc);
        *(uint4*)(vtbf + ((long)bIdx * 256 + c) * 2048 + nLoc0) = u;
    }
}

// ---------------------------------------------------------------------------
// Kernel 2 (per iter): q = LN(slots)@Wq + bq ; also zero uacc/csum for batch.
// Grid: 32 blocks (one batch) x 256 threads.
// ---------------------------------------------------------------------------
__global__ __launch_bounds__(256) void k_q(
    const float* __restrict__ slots, const float* __restrict__ ln_g,
    const float* __restrict__ ln_b,
    const unsigned short* __restrict__ Wqt, const float* __restrict__ bq,
    unsigned short* __restrict__ qbf, float* __restrict__ uacc,
    float* __restrict__ csum)
{
    __shared__ unsigned short al[16 * 256];
    __shared__ float mu[16], rs[16];
    const int tid = threadIdx.x, lane = tid & 31, w = tid >> 5;
    const int half = lane >> 4, nIdx = lane & 15;
    const int b = blockIdx.x;

    for (int i = tid; i < 4096; i += 256) uacc[(long)b * 4096 + i] = 0.0f;
    if (tid < 16) csum[b * 16 + tid] = 0.0f;

    const float* sl = slots + (long)b * 4096;
    if (tid < 16) {
        float s = 0.f, q = 0.f;
        for (int d = 0; d < 256; ++d) { float x = sl[tid * 256 + d]; s += x; q += x * x; }
        float m = s * (1.0f / 256.0f);
        mu[tid] = m; rs[tid] = rsqrtf(q * (1.0f / 256.0f) - m * m + 1e-5f);
    }
    __syncthreads();
    for (int i = tid; i < 4096; i += 256) {
        int r = i >> 8, d = i & 255;
        al[i] = f2bf((sl[i] - mu[r]) * rs[r] * ln_g[d] + ln_b[d]);
    }
    __syncthreads();

    v8f acc[2] = { vzero(), vzero() };
    for (int kb = 0; kb < 256; kb += 32) {
        v16bf a = frag_a_bf16(al, 0, 256, kb);
#pragma unroll
        for (int t = 0; t < 2; ++t)
            acc[t] = wmma_bf16(a, frag_b(Wqt, (2 * w + t) * 16, 256, kb), acc[t]);
    }
#pragma unroll
    for (int t = 0; t < 2; ++t) {
        int c = (2 * w + t) * 16 + nIdx;
        float bc = bq[c];
#pragma unroll
        for (int i = 0; i < 8; ++i)
            qbf[(long)b * 4096 + (8 * half + i) * 256 + c] = f2bf(acc[t][i] + bc);
    }
}

// ---------------------------------------------------------------------------
// Kernel 3 (per iter): fused attention chunk.
//  logits = k@q^T * scale ; row softmax(+1e-8) ; colsum += ; uacc += a^T @ v
// Grid: B*8 = 256 blocks (each = one batch x 256 n-rows), 256 threads.
// ---------------------------------------------------------------------------
__global__ __launch_bounds__(256) void k_attn(
    const unsigned short* __restrict__ kbf, const unsigned short* __restrict__ qbf,
    const unsigned short* __restrict__ vtbf,
    float* __restrict__ uacc, float* __restrict__ csum)
{
    __shared__ unsigned short aT[16 * 256];   // a^T: [s][n_local] bf16
    __shared__ float cpart[16][17];
    const int tid = threadIdx.x, lane = tid & 31, w = tid >> 5;
    const int half = lane >> 4, nIdx = lane & 15;
    const int b = blockIdx.x >> 3, chunk = blockIdx.x & 7, n0 = chunk * 256;
    const unsigned short* kB = kbf + (long)b * 2048 * 256;
    const unsigned short* qB = qbf + (long)b * 16 * 256;
    const unsigned short* vB = vtbf + (long)b * 256 * 2048;

    // logits: wave w owns n row-tiles 2w, 2w+1
    v8f lg[2] = { vzero(), vzero() };
    for (int kb = 0; kb < 256; kb += 32) {
        v16bf bq = frag_b(qB, 0, 256, kb);    // B = q^T (cols = 16 slots)
#pragma unroll
        for (int r = 0; r < 2; ++r) {
            v16bf a = frag_a_bf16(kB, n0 + (2 * w + r) * 16, 256, kb);
            lg[r] = wmma_bf16(a, bq, lg[r]);
        }
    }
    const float scale = 0.0625f;  // 256^-0.5
#pragma unroll
    for (int r = 0; r < 2; ++r)
#pragma unroll
        for (int i = 0; i < 8; ++i) {
            float x = lg[r][i] * scale;
            float m = x;
            m = fmaxf(m, __shfl_xor(m, 1)); m = fmaxf(m, __shfl_xor(m, 2));
            m = fmaxf(m, __shfl_xor(m, 4)); m = fmaxf(m, __shfl_xor(m, 8));
            float e = __expf(x - m);
            float s = e;
            s += __shfl_xor(s, 1); s += __shfl_xor(s, 2);
            s += __shfl_xor(s, 4); s += __shfl_xor(s, 8);
            float av = e / s + 1e-8f;
            aT[nIdx * 256 + (2 * w + r) * 16 + 8 * half + i] = f2bf(av);
        }
    __syncthreads();
    // colsum partials
    {
        int s_ = tid & 15, seg = tid >> 4;
        float sm = 0.f;
        for (int j = 0; j < 16; ++j) sm += bf2f(aT[s_ * 256 + seg * 16 + j]);
        cpart[seg][s_] = sm;
    }
    __syncthreads();
    if (tid < 16) {
        float tot = 0.f;
        for (int seg = 0; seg < 16; ++seg) tot += cpart[seg][tid];
        atomicAdd(&csum[b * 16 + tid], tot);
    }
    // updates partial: u' = a^T @ v  (M=16, N=256, K=256 chunk)
    v8f up[2] = { vzero(), vzero() };
    for (int kb = 0; kb < 256; kb += 32) {
        v16bf a = frag_a_bf16(aT, 0, 256, kb);
#pragma unroll
        for (int t = 0; t < 2; ++t)
            up[t] = wmma_bf16(a, frag_b(vB, (2 * w + t) * 16, 2048, n0 + kb), up[t]);
    }
#pragma unroll
    for (int t = 0; t < 2; ++t) {
        int d = (2 * w + t) * 16 + nIdx;
#pragma unroll
        for (int i = 0; i < 8; ++i) {
            int s_ = 8 * half + i;
            atomicAdd(&uacc[((long)b * 16 + s_) * 256 + d], up[t][i]);
        }
    }
}

// ---------------------------------------------------------------------------
// Kernel 4 (per iter): GRU gates + residual MLP; updates slots in place.
// Grid: 32 blocks (one batch = 16 slot rows), 256 threads, 64KB aliased LDS.
// ---------------------------------------------------------------------------
__global__ __launch_bounds__(256) void k_update(
    float* __restrict__ slots, const float* __restrict__ uacc,
    const float* __restrict__ csum,
    const unsigned short* __restrict__ Wih,   // [768][256] bf16 (row-major == W_ih^T layout)
    const unsigned short* __restrict__ Whh,   // [768][256] bf16
    const float* __restrict__ b_ih, const float* __restrict__ b_hh,
    const float* __restrict__ ln_g, const float* __restrict__ ln_b,
    const unsigned short* __restrict__ W1t, const float* __restrict__ b1,  // [1024][256]
    const unsigned short* __restrict__ W2t, const float* __restrict__ b2)  // [256][1024]
{
    __shared__ __align__(16) char smem[65536];
    float* Grz = (float*)smem;                        // [16][512] r|z sums (gi+gh)
    float* Gni = (float*)(smem + 32768);              // [16][256] gi_n ; later S_new
    float* Gnh = (float*)(smem + 49152);              // [16][256] gh_n
    unsigned short* Ah = (unsigned short*)(smem + 49152);  // [16][256] bf16 (alias)
    float* stMu = (float*)(smem + 57344);
    float* stRs = (float*)(smem + 57408);
    unsigned short* At = (unsigned short*)smem;       // [16][1024] bf16 (alias Grz)
    float* Snew = Gni;

    const int tid = threadIdx.x, lane = tid & 31, w = tid >> 5;
    const int half = lane >> 4, nIdx = lane & 15;
    const int b = blockIdx.x;
    const float csInv = 1.0f / csum[b * 16 + nIdx];   // per-lane row scale

    // warm L2 for the MLP weights
    block_prefetch(W1t, 1024 * 256 * 2);
    block_prefetch(W2t, 256 * 1024 * 2);

    // GEMM1: gi = updates@Wih^T, gh = prev@Whh^T  (M=16, N=768, K=256)
    v8f gi[6], gh[6];
#pragma unroll
    for (int t = 0; t < 6; ++t) { gi[t] = vzero(); gh[t] = vzero(); }
    for (int kb = 0; kb < 256; kb += 32) {
        v16bf au = frag_a_f32(uacc,  (long)b * 16, 256, kb, csInv);
        v16bf ap = frag_a_f32(slots, (long)b * 16, 256, kb, 1.0f);
#pragma unroll
        for (int t = 0; t < 6; ++t) {
            int c0 = (w * 6 + t) * 16;
            gi[t] = wmma_bf16(au, frag_b(Wih, c0, 256, kb), gi[t]);
            gh[t] = wmma_bf16(ap, frag_b(Whh, c0, 256, kb), gh[t]);
        }
    }
#pragma unroll
    for (int t = 0; t < 6; ++t) {
        int c = (w * 6 + t) * 16 + nIdx;
        float bi = b_ih[c], bh = b_hh[c];
#pragma unroll
        for (int i = 0; i < 8; ++i) {
            int s_ = 8 * half + i;
            float vi = gi[t][i] + bi, vh = gh[t][i] + bh;
            if (c < 512) Grz[s_ * 512 + c] = vi + vh;
            else { Gni[s_ * 256 + (c - 512)] = vi; Gnh[s_ * 256 + (c - 512)] = vh; }
        }
    }
    __syncthreads();
    // gates -> S_new (in place over Gni)
    for (int i = tid; i < 4096; i += 256) {
        int s_ = i >> 8, d = i & 255;
        float r = 1.0f / (1.0f + __expf(-Grz[s_ * 512 + d]));
        float z = 1.0f / (1.0f + __expf(-Grz[s_ * 512 + 256 + d]));
        float n = tanhf(Gni[i] + r * Gnh[i]);
        float pv = slots[(long)b * 4096 + i];
        Snew[i] = (1.0f - z) * n + z * pv;
    }
    __syncthreads();
    if (tid < 16) {
        float s = 0.f, q = 0.f;
        for (int d = 0; d < 256; ++d) { float x = Snew[tid * 256 + d]; s += x; q += x * x; }
        float m = s * (1.0f / 256.0f);
        stMu[tid] = m; stRs[tid] = rsqrtf(q * (1.0f / 256.0f) - m * m + 1e-5f);
    }
    __syncthreads();
    for (int i = tid; i < 4096; i += 256) {
        int s_ = i >> 8, d = i & 255;
        Ah[i] = f2bf((Snew[i] - stMu[s_]) * stRs[s_] * ln_g[d] + ln_b[d]);
    }
    __syncthreads();
    // MLP1: t = gelu(h @ W1 + b1)  (M=16, N=1024, K=256)
    v8f t1[8];
#pragma unroll
    for (int t = 0; t < 8; ++t) t1[t] = vzero();
    for (int kb = 0; kb < 256; kb += 32) {
        v16bf a = frag_a_bf16(Ah, 0, 256, kb);
#pragma unroll
        for (int t = 0; t < 8; ++t)
            t1[t] = wmma_bf16(a, frag_b(W1t, (w * 8 + t) * 16, 256, kb), t1[t]);
    }
#pragma unroll
    for (int t = 0; t < 8; ++t) {
        int c = (w * 8 + t) * 16 + nIdx;
        float bc = b1[c];
#pragma unroll
        for (int i = 0; i < 8; ++i) {
            float v = t1[t][i] + bc;
            v = 0.5f * v * (1.0f + erff(v * 0.70710678118f));  // exact gelu
            At[(8 * half + i) * 1024 + c] = f2bf(v);
        }
    }
    __syncthreads();
    // MLP2: y = t @ W2 + b2 ; slots = S_new + y  (M=16, N=256, K=1024)
    v8f t2[2] = { vzero(), vzero() };
    for (int kb = 0; kb < 1024; kb += 32) {
        v16bf a = frag_a_bf16(At, 0, 1024, kb);
#pragma unroll
        for (int t = 0; t < 2; ++t)
            t2[t] = wmma_bf16(a, frag_b(W2t, (2 * w + t) * 16, 1024, kb), t2[t]);
    }
#pragma unroll
    for (int t = 0; t < 2; ++t) {
        int c = (2 * w + t) * 16 + nIdx;
        float bc = b2[c];
#pragma unroll
        for (int i = 0; i < 8; ++i) {
            int s_ = 8 * half + i;
            slots[(long)b * 4096 + s_ * 256 + c] = Snew[s_ * 256 + c] + t2[t][i] + bc;
        }
    }
}

// ---------------------------------------------------------------------------
// Host launcher
// ---------------------------------------------------------------------------
extern "C" void kernel_launch(void* const* d_in, const int* in_sizes, int n_in,
                              void* d_out, int out_size, void* d_ws, size_t ws_size,
                              hipStream_t stream) {
    (void)in_sizes; (void)n_in; (void)out_size; (void)ws_size;
    const float* inputs      = (const float*)d_in[0];
    const float* noise       = (const float*)d_in[1];
    const float* slots_mu    = (const float*)d_in[2];
    const float* slots_sigma = (const float*)d_in[3];
    const float* W_in        = (const float*)d_in[4];
    const float* b_in        = (const float*)d_in[5];
    const float* ln_in_g     = (const float*)d_in[6];
    const float* ln_in_b     = (const float*)d_in[7];
    const float* Wk          = (const float*)d_in[8];
    const float* bk          = (const float*)d_in[9];
    const float* Wq          = (const float*)d_in[10];
    const float* bq          = (const float*)d_in[11];
    const float* Wv          = (const float*)d_in[12];
    const float* bv          = (const float*)d_in[13];
    const float* W_ih        = (const float*)d_in[14];
    const float* b_ih        = (const float*)d_in[15];
    const float* W_hh        = (const float*)d_in[16];
    const float* b_hh        = (const float*)d_in[17];
    const float* ln_s_g      = (const float*)d_in[18];
    const float* ln_s_b      = (const float*)d_in[19];
    const float* ln_m_g      = (const float*)d_in[20];
    const float* ln_m_b      = (const float*)d_in[21];
    const float* W1          = (const float*)d_in[22];
    const float* b1          = (const float*)d_in[23];
    const float* W2          = (const float*)d_in[24];
    const float* b2          = (const float*)d_in[25];

    char* ws = (char*)d_ws;
    size_t off = 0;
    auto nxt = [&](size_t bytes) { char* p = ws + off; off += (bytes + 255) & ~(size_t)255; return p; };
    unsigned short* kbf  = (unsigned short*)nxt((size_t)32 * 2048 * 256 * 2);
    unsigned short* vtbf = (unsigned short*)nxt((size_t)32 * 256 * 2048 * 2);
    unsigned short* qbf  = (unsigned short*)nxt((size_t)32 * 16 * 256 * 2);
    float*          uacc = (float*)nxt((size_t)32 * 16 * 256 * 4);
    float*          csum = (float*)nxt((size_t)32 * 16 * 4);
    unsigned short* Wint = (unsigned short*)nxt((size_t)512 * 256 * 2);
    unsigned short* Wkt  = (unsigned short*)nxt((size_t)256 * 256 * 2);
    unsigned short* Wqt  = (unsigned short*)nxt((size_t)256 * 256 * 2);
    unsigned short* Wvt  = (unsigned short*)nxt((size_t)256 * 256 * 2);
    unsigned short* Wihb = (unsigned short*)nxt((size_t)768 * 256 * 2);
    unsigned short* Whhb = (unsigned short*)nxt((size_t)768 * 256 * 2);
    unsigned short* W1t  = (unsigned short*)nxt((size_t)256 * 1024 * 2);
    unsigned short* W2t  = (unsigned short*)nxt((size_t)1024 * 256 * 2);

    float* slots = (float*)d_out;  // slots live in d_out [32][16][256] f32

    // weight conversions (bf16, transposed where needed for B-fragment loads)
    k_wconv<<<512, 256, 0, stream>>>(W_in, Wint, 512, 256, 1);
    k_wconv<<<256, 256, 0, stream>>>(Wk, Wkt, 256, 256, 1);
    k_wconv<<<256, 256, 0, stream>>>(Wq, Wqt, 256, 256, 1);
    k_wconv<<<256, 256, 0, stream>>>(Wv, Wvt, 256, 256, 1);
    k_wconv<<<768, 256, 0, stream>>>(W_ih, Wihb, 768, 256, 0);
    k_wconv<<<768, 256, 0, stream>>>(W_hh, Whhb, 768, 256, 0);
    k_wconv<<<1024, 256, 0, stream>>>(W1, W1t, 256, 1024, 1);
    k_wconv<<<1024, 256, 0, stream>>>(W2, W2t, 1024, 256, 1);

    k_slots_init<<<512, 256, 0, stream>>>(slots_mu, slots_sigma, noise, slots);
    k_xkv<<<1024, 256, 0, stream>>>(inputs, b_in, ln_in_g, ln_in_b, Wint, Wkt, Wvt,
                                    bk, bv, kbf, vtbf);
    for (int it = 0; it < 3; ++it) {
        k_q<<<32, 256, 0, stream>>>(slots, ln_s_g, ln_s_b, Wqt, bq, qbf, uacc, csum);
        k_attn<<<256, 256, 0, stream>>>(kbf, qbf, vtbf, uacc, csum);
        k_update<<<32, 256, 0, stream>>>(slots, uacc, csum, Wihb, Whhb, b_ih, b_hh,
                                         ln_m_g, ln_m_b, W1t, b1, W2t, b2);
    }
}
